// SCP_Zhang_TF_19344532701423
// MI455X (gfx1250) — compile-verified, matrix operand order checked by reading, not run on hardware
//
#include <hip/hip_runtime.h>

// ---------- types ----------
typedef __attribute__((ext_vector_type(16))) __bf16 v16bf;
typedef __attribute__((ext_vector_type(8)))  float  v8f;
typedef unsigned short u16;
typedef __attribute__((ext_vector_type(8)))  u16 v8us;
typedef __attribute__((ext_vector_type(16))) u16 v16us;

// ---------- helpers ----------
__device__ __forceinline__ u16 f2bf(float f) {
  union { float f; unsigned u; } x; x.f = f;
  return (u16)((x.u + 0x7FFFu + ((x.u >> 16) & 1u)) >> 16);
}

// A/B fragment for v_wmma_*_16x16x32_bf16 (16-bit A 16x32 layout):
// lane<16 holds K = {k..k+7, k+16..k+23}; lane>=16 holds {k+8..k+15, k+24..k+31}
// caller passes p already offset by row*ld + kbase + ((lane>>4)<<3)
__device__ __forceinline__ v16bf load_frag16(const u16* p) {
  v8us lo = *(const v8us*)p;
  v8us hi = *(const v8us*)(p + 16);
  v16us w = __builtin_shufflevector(lo, hi, 0,1,2,3,4,5,6,7,8,9,10,11,12,13,14,15);
  return __builtin_bit_cast(v16bf, w);
}

__device__ __forceinline__ float sigm(float x) { return 1.0f / (1.0f + __expf(-x)); }
__device__ __forceinline__ float ftanh(float x) {
  float e = __expf(2.0f * x);
  return (e - 1.0f) / (e + 1.0f);
}

// ---------- fused LSTM step: g = [A1|A2] @ Wp^T + bias; gate math; h,c update ----------
// A1: [M,K1] bf16 (row stride lda1), A2: [M,K2] bf16 (row stride lda2)
// Wp: [1024, K1+K2] bf16 packed [Wih|Whh], rows = gate bands i,f,g,o of 256
// Cst: [M,256] f32 (in/out), Hout: [M,256] bf16
// grid = (M/128, 256/32), block = 256 (8 waves: 4 M x 2 hidden)
__global__ __launch_bounds__(256) void lstm_fused_kernel(
    const u16* __restrict__ A1, long lda1, int K1,
    const u16* __restrict__ A2, long lda2, int K2,
    const u16* __restrict__ Wp,
    const float* __restrict__ bias,
    float* __restrict__ Cst,
    u16* __restrict__ Hout)
{
  const int K = K1 + K2;
  const int lane = threadIdx.x & 31;
  const int wid  = threadIdx.x >> 5;
  const int wm = wid & 3;
  const int wn = wid >> 2;
  const int mbase = blockIdx.x * 128 + wm * 32;
  const int hidbase = blockIdx.y * 32 + wn * 16;
  const int lr = lane & 15;
  const int ksel = (lane >> 4) << 3;

  v8f acc[4][2];
#pragma unroll
  for (int g = 0; g < 4; ++g)
#pragma unroll
    for (int mf = 0; mf < 2; ++mf) acc[g][mf] = (v8f)0.0f;

  for (int kc = 0; kc < K; kc += 32) {
    const u16* Ab; long lda; int kk;
    if (kc < K1) { Ab = A1; lda = lda1; kk = kc; }
    else         { Ab = A2; lda = lda2; kk = kc - K1; }
    v16bf a[2];
#pragma unroll
    for (int mf = 0; mf < 2; ++mf)
      a[mf] = load_frag16(Ab + (long)(mbase + mf * 16 + lr) * lda + kk + ksel);
    v16bf b[4];
#pragma unroll
    for (int g = 0; g < 4; ++g)
      b[g] = load_frag16(Wp + (long)(g * 256 + hidbase + lr) * K + kc + ksel);
#pragma unroll
    for (int g = 0; g < 4; ++g)
#pragma unroll
      for (int mf = 0; mf < 2; ++mf)
        acc[g][mf] = __builtin_amdgcn_wmma_f32_16x16x32_bf16(
            false, a[mf], false, b[g], (short)0, acc[g][mf], false, false);
  }

  // epilogue: C/D layout -> vgpr i: lanes0-15 M=i, lanes16-31 M=8+i; N = lane&15
  const int hid = hidbase + lr;
  const float bi  = bias[hid];
  const float bff = bias[hid + 256];
  const float bg  = bias[hid + 512];
  const float bo  = bias[hid + 768];
  const int mtop = (lane >> 4) << 3;
#pragma unroll
  for (int mf = 0; mf < 2; ++mf) {
#pragma unroll
    for (int i = 0; i < 8; ++i) {
      const int m = mbase + mf * 16 + mtop + i;
      const long idx = (long)m * 256 + hid;
      float gi = acc[0][mf][i] + bi;
      float gf = acc[1][mf][i] + bff;
      float gg = acc[2][mf][i] + bg;
      float go = acc[3][mf][i] + bo;
      float c = sigm(gf) * Cst[idx] + sigm(gi) * ftanh(gg);
      Cst[idx] = c;
      Hout[idx] = f2bf(sigm(go) * ftanh(c));
    }
  }
}

// ---------- GEMM + bias (+relu): C = A @ W^T ----------
// A:[M,K] bf16 (lda), W:[N,K] bf16, bias:[N] f32
// Cf (f32, stride ldcf) and/or Cb (bf16, stride ldcb)
// grid = (M/64, ceil(N/256)), block = 256 (8 waves: 2 M x 4 N)
__global__ __launch_bounds__(256) void gemm_bias_act_kernel(
    const u16* __restrict__ A, long lda,
    const u16* __restrict__ W,
    const float* __restrict__ bias,
    float* __restrict__ Cf, long ldcf,
    u16* __restrict__ Cb, long ldcb,
    int N, int K, int relu)
{
  const int lane = threadIdx.x & 31;
  const int wid  = threadIdx.x >> 5;
  const int wm = wid & 1;
  const int wn = wid >> 1;
  const int mbase = blockIdx.x * 64 + wm * 32;
  const int nbase = blockIdx.y * 256 + wn * 64;
  const int lr = lane & 15;
  const int ksel = (lane >> 4) << 3;

  v8f acc[2][4];
#pragma unroll
  for (int mf = 0; mf < 2; ++mf)
#pragma unroll
    for (int nf = 0; nf < 4; ++nf) acc[mf][nf] = (v8f)0.0f;

  for (int kc = 0; kc < K; kc += 32) {
    v16bf a[2];
#pragma unroll
    for (int mf = 0; mf < 2; ++mf)
      a[mf] = load_frag16(A + (long)(mbase + mf * 16 + lr) * lda + kc + ksel);
#pragma unroll
    for (int nf = 0; nf < 4; ++nf) {
      if (nbase + nf * 16 < N) {
        v16bf b = load_frag16(W + (long)(nbase + nf * 16 + lr) * K + kc + ksel);
#pragma unroll
        for (int mf = 0; mf < 2; ++mf)
          acc[mf][nf] = __builtin_amdgcn_wmma_f32_16x16x32_bf16(
              false, a[mf], false, b, (short)0, acc[mf][nf], false, false);
      }
    }
  }

  const int mtop = (lane >> 4) << 3;
#pragma unroll
  for (int nf = 0; nf < 4; ++nf) {
    if (nbase + nf * 16 >= N) continue;
    const int n = nbase + nf * 16 + lr;
    const float bv = bias[n];
#pragma unroll
    for (int mf = 0; mf < 2; ++mf) {
#pragma unroll
      for (int i = 0; i < 8; ++i) {
        const int m = mbase + mf * 16 + mtop + i;
        float v = acc[mf][nf][i] + bv;
        if (relu) v = fmaxf(v, 0.0f);
        if (Cf) Cf[(long)m * ldcf + n] = v;
        if (Cb) Cb[(long)m * ldcb + n] = f2bf(v);
      }
    }
  }
}

// ---------- setup kernels ----------
__global__ void f32_to_bf16_kernel(const float* __restrict__ s, u16* __restrict__ d, long n) {
  long i = (long)blockIdx.x * blockDim.x + threadIdx.x;
  long st = (long)gridDim.x * blockDim.x;
  for (; i < n; i += st) d[i] = f2bf(s[i]);
}

__global__ void pack_w_kernel(const float* __restrict__ Wih, int K1,
                              const float* __restrict__ Whh, int K2,
                              u16* __restrict__ out, long rows) {
  const int K = K1 + K2;
  const long n = rows * (long)K;
  long i = (long)blockIdx.x * blockDim.x + threadIdx.x;
  long st = (long)gridDim.x * blockDim.x;
  for (; i < n; i += st) {
    long r = i / K;
    int k = (int)(i - r * K);
    float v = (k < K1) ? Wih[r * K1 + k] : Whh[r * (long)K2 + (k - K1)];
    out[i] = f2bf(v);
  }
}

__global__ void add_vec_kernel(const float* __restrict__ a, const float* __restrict__ b,
                               float* __restrict__ o, int n) {
  int i = blockIdx.x * blockDim.x + threadIdx.x;
  if (i < n) o[i] = a[i] + b[i];
}

__global__ void zero32_kernel(unsigned* __restrict__ p, long n) {
  long i = (long)blockIdx.x * blockDim.x + threadIdx.x;
  long st = (long)gridDim.x * blockDim.x;
  for (; i < n; i += st) p[i] = 0u;
}

// ---------- host orchestration ----------
extern "C" void kernel_launch(void* const* d_in, const int* in_sizes, int n_in,
                              void* d_out, int out_size, void* d_ws, size_t ws_size,
                              hipStream_t stream) {
  const int Dd = 64, Hh = 256, DN = 512, Bn = 2048, T = 168, WOUT = 24;

  const float* x_in   = (const float*)d_in[0];
  const float* eWih0  = (const float*)d_in[1];
  const float* eWhh0  = (const float*)d_in[2];
  const float* ebih0  = (const float*)d_in[3];
  const float* ebhh0  = (const float*)d_in[4];
  const float* eWih1  = (const float*)d_in[5];
  const float* eWhh1  = (const float*)d_in[6];
  const float* ebih1  = (const float*)d_in[7];
  const float* ebhh1  = (const float*)d_in[8];
  const float* dWih0  = (const float*)d_in[9];
  const float* dWhh0  = (const float*)d_in[10];
  const float* dbih0  = (const float*)d_in[11];
  const float* dbhh0  = (const float*)d_in[12];
  const float* dWih1  = (const float*)d_in[13];
  const float* dWhh1  = (const float*)d_in[14];
  const float* dbih1  = (const float*)d_in[15];
  const float* dbhh1  = (const float*)d_in[16];
  const float* fc1_w  = (const float*)d_in[17];
  const float* fc1_b  = (const float*)d_in[18];
  const float* fc2_w  = (const float*)d_in[19];
  const float* fc2_b  = (const float*)d_in[20];
  float* out = (float*)d_out;

  // bump allocator on workspace
  char* ws = (char*)d_ws;
  size_t off = 0;
  auto alloc = [&](size_t bytes) -> char* {
    char* p = ws + off;
    off += (bytes + 255) & ~(size_t)255;
    return p;
  };

  u16* x_bf   = (u16*)alloc((size_t)Bn * T * Dd * 2);
  u16* encW0p = (u16*)alloc((size_t)1024 * (Dd + Hh) * 2);
  u16* encW1p = (u16*)alloc((size_t)1024 * (Hh + Hh) * 2);
  u16* decW0p = (u16*)alloc((size_t)1024 * (Dd + Hh) * 2);
  u16* decW1p = (u16*)alloc((size_t)1024 * (Hh + Hh) * 2);
  u16* fc1_bf = (u16*)alloc((size_t)DN * Hh * 2);
  u16* fc2_bf = (u16*)alloc((size_t)Dd * DN * 2);
  float* db_e0 = (float*)alloc(1024 * 4);
  float* db_e1 = (float*)alloc(1024 * 4);
  float* db_d0 = (float*)alloc(1024 * 4);
  float* db_d1 = (float*)alloc(1024 * 4);
  u16* h0[2] = { (u16*)alloc((size_t)Bn * Hh * 2), (u16*)alloc((size_t)Bn * Hh * 2) };
  u16* h1[2] = { (u16*)alloc((size_t)Bn * Hh * 2), (u16*)alloc((size_t)Bn * Hh * 2) };
  float* c0 = (float*)alloc((size_t)Bn * Hh * 4);
  float* c1 = (float*)alloc((size_t)Bn * Hh * 4);
  u16* z_bf  = (u16*)alloc((size_t)Bn * Dd * 2);
  u16* y1_bf = (u16*)alloc((size_t)Bn * DN * 2);

  // ---- setup (repeated every call: deterministic) ----
  f32_to_bf16_kernel<<<1024, 256, 0, stream>>>(x_in, x_bf, (long)Bn * T * Dd);
  pack_w_kernel<<<512, 256, 0, stream>>>(eWih0, Dd, eWhh0, Hh, encW0p, 1024);
  pack_w_kernel<<<1024, 256, 0, stream>>>(eWih1, Hh, eWhh1, Hh, encW1p, 1024);
  pack_w_kernel<<<512, 256, 0, stream>>>(dWih0, Dd, dWhh0, Hh, decW0p, 1024);
  pack_w_kernel<<<1024, 256, 0, stream>>>(dWih1, Hh, dWhh1, Hh, decW1p, 1024);
  f32_to_bf16_kernel<<<512, 256, 0, stream>>>(fc1_w, fc1_bf, (long)DN * Hh);
  f32_to_bf16_kernel<<<128, 256, 0, stream>>>(fc2_w, fc2_bf, (long)Dd * DN);
  add_vec_kernel<<<4, 256, 0, stream>>>(ebih0, ebhh0, db_e0, 1024);
  add_vec_kernel<<<4, 256, 0, stream>>>(ebih1, ebhh1, db_e1, 1024);
  add_vec_kernel<<<4, 256, 0, stream>>>(dbih0, dbhh0, db_d0, 1024);
  add_vec_kernel<<<4, 256, 0, stream>>>(dbih1, dbhh1, db_d1, 1024);
  zero32_kernel<<<1024, 256, 0, stream>>>((unsigned*)h0[0], (long)Bn * Hh / 2);
  zero32_kernel<<<1024, 256, 0, stream>>>((unsigned*)h1[0], (long)Bn * Hh / 2);
  zero32_kernel<<<2048, 256, 0, stream>>>((unsigned*)c0, (long)Bn * Hh);
  zero32_kernel<<<2048, 256, 0, stream>>>((unsigned*)c1, (long)Bn * Hh);

  const dim3 lstm_grid(Bn / 128, 8);   // 256 hidden / 32
  int p0 = 0, p1 = 0;

  // ---- encoder: layer0 + layer1 fused per time step ----
  for (int t = 0; t < T; ++t) {
    lstm_fused_kernel<<<lstm_grid, 256, 0, stream>>>(
        x_bf + (long)t * Dd, (long)T * Dd, Dd,
        h0[p0], Hh, Hh, encW0p, db_e0, c0, h0[p0 ^ 1]);
    p0 ^= 1;
    lstm_fused_kernel<<<lstm_grid, 256, 0, stream>>>(
        h0[p0], Hh, Hh,
        h1[p1], Hh, Hh, encW1p, db_e1, c1, h1[p1 ^ 1]);
    p1 ^= 1;
  }

  // ---- decoder: 2 cells + MLP head per step, z feedback ----
  for (int s = 0; s < WOUT; ++s) {
    const u16* zin = (s == 0) ? (x_bf + (long)(T - 1) * Dd) : z_bf;
    const long zlda = (s == 0) ? (long)T * Dd : (long)Dd;
    lstm_fused_kernel<<<lstm_grid, 256, 0, stream>>>(
        zin, zlda, Dd,
        h0[p0], Hh, Hh, decW0p, db_d0, c0, h0[p0 ^ 1]);
    p0 ^= 1;
    lstm_fused_kernel<<<lstm_grid, 256, 0, stream>>>(
        h0[p0], Hh, Hh,
        h1[p1], Hh, Hh, decW1p, db_d1, c1, h1[p1 ^ 1]);
    p1 ^= 1;
    // fc1 + relu -> y1 (bf16)
    gemm_bias_act_kernel<<<dim3(Bn / 64, (DN + 255) / 256), 256, 0, stream>>>(
        h1[p1], Hh, fc1_bf, fc1_b,
        nullptr, 0, y1_bf, DN, DN, Hh, 1);
    // fc2 -> out[:, s, :] (f32) and z feedback (bf16)
    gemm_bias_act_kernel<<<dim3(Bn / 64, 1), 256, 0, stream>>>(
        y1_bf, DN, fc2_bf, fc2_b,
        out + (long)s * Dd, (long)WOUT * Dd, z_bf, Dd, Dd, DN, 0);
  }
}
